// SimpleS5Block_85229331022410
// MI455X (gfx1250) — compile-verified, matrix operand order checked by reading, not run on hardware
//
#include <hip/hip_runtime.h>
#include <hip/hip_bf16.h>
#include <math.h>

// ---------------------------------------------------------------------------
// S5Block on MI455X (gfx1250): LN1 -> S5 (WMMA GEMMs + LDS chunked scan)
// -> GELU+residual -> LN2 -> GEGLU FF -> residual.
// Big GEMMs stage A-tiles via GLOBAL_LOAD_ASYNC_TO_LDS (ASYNCcnt) and
// weight tiles via the Tensor Data Mover (TENSORcnt), double buffered in LDS,
// feeding v_wmma_f32_16x16x32_bf16.
// ---------------------------------------------------------------------------

typedef __attribute__((ext_vector_type(16))) __bf16 v16bf;
typedef __attribute__((ext_vector_type(8)))  __bf16 v8bf;
typedef __attribute__((ext_vector_type(8)))  float  v8f;
typedef __attribute__((ext_vector_type(4)))  unsigned u32x4;
typedef __attribute__((ext_vector_type(8)))  unsigned u32x8;

constexpr int BATCH = 4;
constexpr int LSEQ  = 2048;
constexpr int DDIM  = 1024;
constexpr int PDIM  = 64;
constexpr int P2    = 128;    // [re | im]
constexpr int FDIM  = 2048;   // F
constexpr int F2    = 4096;   // 2F
constexpr int BL    = BATCH * LSEQ;   // 8192 rows

__device__ __forceinline__ float gelu_exact(float x) {
    return 0.5f * x * (1.0f + erff(x * 0.70710678118654752440f));
}

// ---------------------------------------------------------------------------
// WMMA fragment helpers (16x16x32 bf16, wave32)
// ---------------------------------------------------------------------------
__device__ __forceinline__ v16bf load_frag16x32(const __bf16* base, int ld, int lane) {
    const int row = lane & 15;
    const int hi  = (lane >> 4) << 3;   // 0 or 8
    const __bf16* p = base + (size_t)row * ld + hi;
    union { v16bf v; v8bf h[2]; } u;
    u.h[0] = *(const v8bf*)(p);
    u.h[1] = *(const v8bf*)(p + 16);
    return u.v;
}

// LDS-tile variant (row stride 32 elements) -> ds_load_b128 pairs
__device__ __forceinline__ v16bf load_frag_lds(const __bf16* base, int lane) {
    const int row = lane & 15;
    const int hi  = (lane >> 4) << 3;
    const __bf16* p = base + row * 32 + hi;
    union { v16bf v; v8bf h[2]; } u;
    u.h[0] = *(const v8bf*)(p);
    u.h[1] = *(const v8bf*)(p + 16);
    return u.v;
}

__device__ __forceinline__ void store_tile_f32(float* C, int ldc, int lane, const v8f& acc) {
    const int col   = lane & 15;
    const int rbase = (lane >> 4) << 3;
#pragma unroll
    for (int r = 0; r < 8; ++r)
        C[(size_t)(rbase + r) * ldc + col] = acc[r];
}

// ---------------------------------------------------------------------------
// Async staging of a 128x32 bf16 activation tile into LDS (8KB).
// 512 x 16B transfers: 2 per thread, tracked on ASYNCcnt.
// ---------------------------------------------------------------------------
__device__ __forceinline__ void stage_a_async(const __bf16* gsrc, int lda,
                                              const __bf16* lds_tile, int tid) {
#pragma unroll
    for (int q = 0; q < 2; ++q) {
        int t   = tid + q * 256;
        int row = t >> 2;
        int seg = t & 3;
        const __bf16* gp = gsrc + (size_t)row * lda + seg * 8;     // 16B chunk
        unsigned lo = (unsigned)(uintptr_t)(lds_tile + row * 32 + seg * 8);
        asm volatile("global_load_async_to_lds_b128 %0, %1, off"
                     :: "v"(lo), "v"(gp) : "memory");
    }
}

// ---------------------------------------------------------------------------
// TDM: DMA a 2D weight tile (tile_dim0=32 elems, tile_dim1=128 rows,
// data_size=2B, row stride = ld elems) from global into LDS. Scalar D#.
// ---------------------------------------------------------------------------
__device__ __forceinline__ void tdm_load_tile(const __bf16* gsrc, int ld, int rows,
                                              const __bf16* lds_tile) {
    unsigned long long ga = (unsigned long long)(uintptr_t)gsrc;
    u32x4 g0;
    g0[0] = 1u;                                            // count=1 (valid user D#)
    g0[1] = (unsigned)(uintptr_t)lds_tile;                 // lds_addr (bytes)
    g0[2] = (unsigned)(ga & 0xFFFFFFFFu);                  // global_addr[31:0]
    g0[3] = (unsigned)((ga >> 32) & 0x1FFFFFFu) | 0x80000000u; // ga[56:32] | type=2
    u32x8 g1;
    g1[0] = 1u << 16;                                      // data_size = 2 bytes
    g1[1] = ((unsigned)ld & 0xFFFFu) << 16;                // tensor_dim0[15:0]
    g1[2] = ((unsigned)ld >> 16) | (((unsigned)rows & 0xFFFFu) << 16); // dim0 hi | dim1 lo
    g1[3] = ((unsigned)rows >> 16) | (32u << 16);          // dim1 hi | tile_dim0 = 32
    g1[4] = 128u;                                          // tile_dim1 = 128 (tile_dim2=0)
    g1[5] = (unsigned)ld;                                  // tensor_dim0_stride[31:0]
    g1[6] = 0u;                                            // stride hi | dim1_stride lo
    g1[7] = 0u;
    asm volatile("tensor_load_to_lds %0, %1" :: "s"(g0), "s"(g1) : "memory");
}

// ---------------------------------------------------------------------------
// Kernel: SSM setup — ZOH discretization.
// ---------------------------------------------------------------------------
__global__ void ssm_setup_kernel(const float* __restrict__ lre,
                                 const float* __restrict__ lim,
                                 const float* __restrict__ log_step,
                                 float* __restrict__ Are, float* __restrict__ Aim,
                                 float* __restrict__ Kre, float* __restrict__ Kim) {
    int p = threadIdx.x;
    if (p >= PDIM) return;
    float lr = lre[p], li = lim[p];
    float dt = expf(log_step[p]);
    float er = expf(lr * dt);
    float th = li * dt;
    float ar = er * cosf(th);
    float ai = er * sinf(th);
    Are[p] = ar; Aim[p] = ai;
    float nr = ar - 1.0f, ni = ai;
    float den = lr * lr + li * li;
    Kre[p] = (nr * lr + ni * li) / den;
    Kim[p] = (ni * lr - nr * li) / den;
}

__global__ void pack_bbar_kernel(const float* __restrict__ Bre,
                                 const float* __restrict__ Bim,
                                 const float* __restrict__ Kre,
                                 const float* __restrict__ Kim,
                                 __bf16* __restrict__ out) {
    int idx = blockIdx.x * 256 + threadIdx.x;   // over P*D
    int p = idx >> 10;
    int d = idx & 1023;
    float br = Bre[idx], bi = Bim[idx];
    float cr = Kre[p],   ci = Kim[p];
    out[(size_t)p * DDIM + d]          = (__bf16)(cr * br - ci * bi);
    out[(size_t)(PDIM + p) * DDIM + d] = (__bf16)(cr * bi + ci * br);
}

__global__ void pack_c_kernel(const float* __restrict__ Cre,
                              const float* __restrict__ Cim,
                              __bf16* __restrict__ out) {
    int idx = blockIdx.x * 256 + threadIdx.x;   // over D*P
    int d = idx >> 6;
    int p = idx & 63;
    out[(size_t)d * P2 + p]        = (__bf16)( 2.0f * Cre[idx]);
    out[(size_t)d * P2 + PDIM + p] = (__bf16)(-2.0f * Cim[idx]);
}

__global__ void cvt_bf16_kernel(const float* __restrict__ in,
                                __bf16* __restrict__ out, long n) {
    long i = (long)blockIdx.x * 256 + threadIdx.x;
    if (i < n) out[i] = (__bf16)in[i];
}

// ---------------------------------------------------------------------------
// LayerNorm over D=1024
// ---------------------------------------------------------------------------
__global__ void ln_kernel(const float* __restrict__ x,
                          const float* __restrict__ g,
                          const float* __restrict__ b,
                          float* __restrict__ outf,
                          __bf16* __restrict__ outbf) {
    __shared__ float s1[256], s2[256];
    const size_t base = (size_t)blockIdx.x * DDIM;
    const int tid = threadIdx.x;
    float xv[4], s = 0.f, ss = 0.f;
#pragma unroll
    for (int k = 0; k < 4; ++k) {
        float v = x[base + tid + k * 256];
        xv[k] = v; s += v; ss += v * v;
    }
    s1[tid] = s; s2[tid] = ss;
    __syncthreads();
    for (int o = 128; o > 0; o >>= 1) {
        if (tid < o) { s1[tid] += s1[tid + o]; s2[tid] += s2[tid + o]; }
        __syncthreads();
    }
    const float mu  = s1[0] * (1.0f / DDIM);
    const float var = s2[0] * (1.0f / DDIM) - mu * mu;
    const float rstd = rsqrtf(var + 1e-5f);
#pragma unroll
    for (int k = 0; k < 4; ++k) {
        int c = tid + k * 256;
        float v = (xv[k] - mu) * rstd * g[c] + b[c];
        outf[base + c]  = v;
        outbf[base + c] = (__bf16)v;
    }
}

// ---------------------------------------------------------------------------
// GEMM 1: Bu = fx_bf (BLx1024) * Bbar^T (128x1024) -> f32 (BLx128)
// ---------------------------------------------------------------------------
__global__ void gemm_bu_kernel(const __bf16* __restrict__ A,
                               const __bf16* __restrict__ Bm,
                               float* __restrict__ C) {
    const int lane = threadIdx.x & 31, w = threadIdx.x >> 5;
    const int wm = w >> 2, wn = w & 3;
    const int m0 = blockIdx.y * 128 + wm * 64;
    const int n0 = blockIdx.x * 128 + wn * 32;
    v8f acc[4][2];
#pragma unroll
    for (int i = 0; i < 4; ++i)
#pragma unroll
        for (int j = 0; j < 2; ++j) acc[i][j] = {};
    const __bf16* Ab = A  + (size_t)m0 * DDIM;
    const __bf16* Bb = Bm + (size_t)n0 * DDIM;
    for (int k = 0; k < DDIM; k += 32) {
        __builtin_prefetch(Ab + k + 64, 0, 3);
        v16bf a[4];
#pragma unroll
        for (int i = 0; i < 4; ++i)
            a[i] = load_frag16x32(Ab + (size_t)(i * 16) * DDIM + k, DDIM, lane);
#pragma unroll
        for (int j = 0; j < 2; ++j) {
            v16bf bf = load_frag16x32(Bb + (size_t)(j * 16) * DDIM + k, DDIM, lane);
#pragma unroll
            for (int i = 0; i < 4; ++i)
                acc[i][j] = __builtin_amdgcn_wmma_f32_16x16x32_bf16(
                    false, a[i], false, bf, (short)0, acc[i][j], false, false);
        }
    }
#pragma unroll
    for (int i = 0; i < 4; ++i)
#pragma unroll
        for (int j = 0; j < 2; ++j)
            store_tile_f32(C + (size_t)(m0 + i * 16) * P2 + (n0 + j * 16), P2, lane, acc[i][j]);
}

// ---------------------------------------------------------------------------
// Chunked parallel scan
// ---------------------------------------------------------------------------
__global__ void scan_kernel(const float* __restrict__ Bu,
                            __bf16* __restrict__ xs,
                            const float* __restrict__ Are,
                            const float* __restrict__ Aim) {
    __shared__ float sr[256], si[256];
    const int b = blockIdx.x / PDIM;
    const int p = blockIdx.x % PDIM;
    const int tid = threadIdx.x;
    const float ar = Are[p], ai = Aim[p];

    const int l0 = tid * 8;
    float xr = 0.f, xi = 0.f;
#pragma unroll
    for (int t = 0; t < 8; ++t) {
        size_t idx = ((size_t)(b * LSEQ + l0 + t)) * P2 + p;
        float br = Bu[idx], bi = Bu[idx + PDIM];
        float nr = ar * xr - ai * xi + br;
        float ni = ar * xi + ai * xr + bi;
        xr = nr; xi = ni;
    }
    sr[tid] = xr; si[tid] = xi;

    float mr = ar, mi = ai;
#pragma unroll
    for (int t = 0; t < 3; ++t) { float tr = mr * mr - mi * mi; mi = 2.f * mr * mi; mr = tr; }

    for (int o = 1; o < 256; o <<= 1) {
        __syncthreads();
        float pr = 0.f, pi = 0.f;
        if (tid >= o) { pr = sr[tid - o]; pi = si[tid - o]; }
        __syncthreads();
        if (tid >= o) {
            sr[tid] += mr * pr - mi * pi;
            si[tid] += mr * pi + mi * pr;
        }
        float tr = mr * mr - mi * mi; mi = 2.f * mr * mi; mr = tr;
    }
    __syncthreads();

    float x0r = (tid > 0) ? sr[tid - 1] : 0.f;
    float x0i = (tid > 0) ? si[tid - 1] : 0.f;
#pragma unroll
    for (int t = 0; t < 8; ++t) {
        size_t idx = ((size_t)(b * LSEQ + l0 + t)) * P2 + p;
        float br = Bu[idx], bi = Bu[idx + PDIM];
        float nr = ar * x0r - ai * x0i + br;
        float ni = ar * x0i + ai * x0r + bi;
        x0r = nr; x0i = ni;
        xs[idx]        = (__bf16)x0r;
        xs[idx + PDIM] = (__bf16)x0i;
    }
}

// ---------------------------------------------------------------------------
// GEMM 2: y = xs (BLx128) * Cpack^T (1024x128); fused Dskip*u + GELU + residual
// ---------------------------------------------------------------------------
__global__ void gemm_y_kernel(const __bf16* __restrict__ A,
                              const __bf16* __restrict__ Bm,
                              const float* __restrict__ Dskip,
                              float* __restrict__ fxh) {
    const int lane = threadIdx.x & 31, w = threadIdx.x >> 5;
    const int wm = w >> 2, wn = w & 3;
    const int m0 = blockIdx.y * 128 + wm * 64;
    const int n0 = blockIdx.x * 128 + wn * 32;
    v8f acc[4][2];
#pragma unroll
    for (int i = 0; i < 4; ++i)
#pragma unroll
        for (int j = 0; j < 2; ++j) acc[i][j] = {};
    const __bf16* Ab = A  + (size_t)m0 * P2;
    const __bf16* Bb = Bm + (size_t)n0 * P2;
#pragma unroll
    for (int k = 0; k < P2; k += 32) {
        v16bf a[4];
#pragma unroll
        for (int i = 0; i < 4; ++i)
            a[i] = load_frag16x32(Ab + (size_t)(i * 16) * P2 + k, P2, lane);
#pragma unroll
        for (int j = 0; j < 2; ++j) {
            v16bf bf = load_frag16x32(Bb + (size_t)(j * 16) * P2 + k, P2, lane);
#pragma unroll
            for (int i = 0; i < 4; ++i)
                acc[i][j] = __builtin_amdgcn_wmma_f32_16x16x32_bf16(
                    false, a[i], false, bf, (short)0, acc[i][j], false, false);
        }
    }
    const int col   = lane & 15;
    const int rbase = (lane >> 4) << 3;
#pragma unroll
    for (int i = 0; i < 4; ++i)
#pragma unroll
        for (int j = 0; j < 2; ++j) {
            int c = n0 + j * 16 + col;
            float dk = Dskip[c];
#pragma unroll
            for (int r = 0; r < 8; ++r) {
                size_t off = (size_t)(m0 + i * 16 + rbase + r) * DDIM + c;
                float f = fxh[off];
                float t = acc[i][j][r] + dk * f;
                fxh[off] = gelu_exact(t) + f;
            }
        }
}

// ---------------------------------------------------------------------------
// GEMM 3 (encoder + fused GEGLU), LDS-staged + double buffered:
// A tiles via async-to-LDS (ASYNCcnt), weight tiles via TDM (TENSORcnt).
// ---------------------------------------------------------------------------
__global__ void gemm_enc_kernel(const __bf16* __restrict__ A,
                                const __bf16* __restrict__ W,
                                __bf16* __restrict__ g) {
    __shared__ __bf16 As[2][128 * 32];
    __shared__ __bf16 Bas[2][128 * 32];
    __shared__ __bf16 Bgs[2][128 * 32];
    const int tid = threadIdx.x;
    const int lane = tid & 31, w = tid >> 5;
    const int wm = w >> 2, wn = w & 3;
    const int mblk = blockIdx.y * 128;
    const int nblk = blockIdx.x * 128;      // column within F
    v8f acca[4][2], accg[4][2];
#pragma unroll
    for (int i = 0; i < 4; ++i)
#pragma unroll
        for (int j = 0; j < 2; ++j) { acca[i][j] = {}; accg[i][j] = {}; }
    const __bf16* Ab = A + (size_t)mblk * DDIM;
    const __bf16* Wa = W + (size_t)nblk * DDIM;            // 'a' half rows
    const __bf16* Wg = W + (size_t)(FDIM + nblk) * DDIM;   // 'gate' half rows

    // prologue: stage k=0 and k=32
#pragma unroll
    for (int s = 0; s < 2; ++s) {
        stage_a_async(Ab + s * 32, DDIM, As[s], tid);
        if (w == 0) {
            tdm_load_tile(Wa + s * 32, DDIM, F2, Bas[s]);
            tdm_load_tile(Wg + s * 32, DDIM, F2, Bgs[s]);
        }
    }

    for (int kk = 0; kk < DDIM; kk += 32) {
        const int buf = (kk >> 5) & 1;
        if (kk + 32 < DDIM) {
            asm volatile("s_wait_asynccnt 0x2" ::: "memory");
            if (w == 0) __builtin_amdgcn_s_wait_tensorcnt(2);
        } else {
            asm volatile("s_wait_asynccnt 0x0" ::: "memory");
            if (w == 0) __builtin_amdgcn_s_wait_tensorcnt(0);
        }
        __syncthreads();

        v16bf a[4];
#pragma unroll
        for (int i = 0; i < 4; ++i)
            a[i] = load_frag_lds(&As[buf][(wm * 64 + i * 16) * 32], lane);
#pragma unroll
        for (int j = 0; j < 2; ++j) {
            v16bf ba = load_frag_lds(&Bas[buf][(wn * 32 + j * 16) * 32], lane);
#pragma unroll
            for (int i = 0; i < 4; ++i)
                acca[i][j] = __builtin_amdgcn_wmma_f32_16x16x32_bf16(
                    false, a[i], false, ba, (short)0, acca[i][j], false, false);
            v16bf bg = load_frag_lds(&Bgs[buf][(wn * 32 + j * 16) * 32], lane);
#pragma unroll
            for (int i = 0; i < 4; ++i)
                accg[i][j] = __builtin_amdgcn_wmma_f32_16x16x32_bf16(
                    false, a[i], false, bg, (short)0, accg[i][j], false, false);
        }
        __syncthreads();

        if (kk + 64 < DDIM) {
            stage_a_async(Ab + kk + 64, DDIM, As[buf], tid);
            if (w == 0) {
                tdm_load_tile(Wa + kk + 64, DDIM, F2, Bas[buf]);
                tdm_load_tile(Wg + kk + 64, DDIM, F2, Bgs[buf]);
            }
        }
    }

    const int col   = lane & 15;
    const int rbase = (lane >> 4) << 3;
#pragma unroll
    for (int i = 0; i < 4; ++i)
#pragma unroll
        for (int j = 0; j < 2; ++j) {
            int c = nblk + wn * 32 + j * 16 + col;
#pragma unroll
            for (int r = 0; r < 8; ++r) {
                size_t off = (size_t)(mblk + wm * 64 + i * 16 + rbase + r) * FDIM + c;
                g[off] = (__bf16)(acca[i][j][r] * gelu_exact(accg[i][j][r]));
            }
        }
}

// ---------------------------------------------------------------------------
// GEMM 4 (decoder): out = g (BLx2048) * Wdec^T (1024x2048) + fx2 -> f32
// Same LDS staging scheme (1 TDM descriptor per stage).
// ---------------------------------------------------------------------------
__global__ void gemm_dec_kernel(const __bf16* __restrict__ A,
                                const __bf16* __restrict__ W,
                                const float* __restrict__ fx2,
                                float* __restrict__ out) {
    __shared__ __bf16 As[2][128 * 32];
    __shared__ __bf16 Bs[2][128 * 32];
    const int tid = threadIdx.x;
    const int lane = tid & 31, w = tid >> 5;
    const int wm = w >> 2, wn = w & 3;
    const int mblk = blockIdx.y * 128;
    const int nblk = blockIdx.x * 128;
    v8f acc[4][2];
#pragma unroll
    for (int i = 0; i < 4; ++i)
#pragma unroll
        for (int j = 0; j < 2; ++j) acc[i][j] = {};
    const __bf16* Ab = A + (size_t)mblk * FDIM;
    const __bf16* Wb = W + (size_t)nblk * FDIM;

#pragma unroll
    for (int s = 0; s < 2; ++s) {
        stage_a_async(Ab + s * 32, FDIM, As[s], tid);
        if (w == 0) tdm_load_tile(Wb + s * 32, FDIM, DDIM, Bs[s]);
    }

    for (int kk = 0; kk < FDIM; kk += 32) {
        const int buf = (kk >> 5) & 1;
        if (kk + 32 < FDIM) {
            asm volatile("s_wait_asynccnt 0x2" ::: "memory");
            if (w == 0) __builtin_amdgcn_s_wait_tensorcnt(1);
        } else {
            asm volatile("s_wait_asynccnt 0x0" ::: "memory");
            if (w == 0) __builtin_amdgcn_s_wait_tensorcnt(0);
        }
        __syncthreads();

        v16bf a[4];
#pragma unroll
        for (int i = 0; i < 4; ++i)
            a[i] = load_frag_lds(&As[buf][(wm * 64 + i * 16) * 32], lane);
#pragma unroll
        for (int j = 0; j < 2; ++j) {
            v16bf bf = load_frag_lds(&Bs[buf][(wn * 32 + j * 16) * 32], lane);
#pragma unroll
            for (int i = 0; i < 4; ++i)
                acc[i][j] = __builtin_amdgcn_wmma_f32_16x16x32_bf16(
                    false, a[i], false, bf, (short)0, acc[i][j], false, false);
        }
        __syncthreads();

        if (kk + 64 < FDIM) {
            stage_a_async(Ab + kk + 64, FDIM, As[buf], tid);
            if (w == 0) tdm_load_tile(Wb + kk + 64, FDIM, DDIM, Bs[buf]);
        }
    }

    const int col   = lane & 15;
    const int rbase = (lane >> 4) << 3;
#pragma unroll
    for (int i = 0; i < 4; ++i)
#pragma unroll
        for (int j = 0; j < 2; ++j) {
            int c = nblk + wn * 32 + j * 16 + col;
#pragma unroll
            for (int r = 0; r < 8; ++r) {
                size_t off = (size_t)(mblk + wm * 64 + i * 16 + rbase + r) * DDIM + c;
                out[off] = acc[i][j][r] + fx2[off];
            }
        }
}

// ---------------------------------------------------------------------------
// Host-side launch
// ---------------------------------------------------------------------------
extern "C" void kernel_launch(void* const* d_in, const int* in_sizes, int n_in,
                              void* d_out, int out_size, void* d_ws, size_t ws_size,
                              hipStream_t stream) {
    const float* x      = (const float*)d_in[0];
    const float* ln1_g  = (const float*)d_in[1];
    const float* ln1_b  = (const float*)d_in[2];
    const float* ln2_g  = (const float*)d_in[3];
    const float* ln2_b  = (const float*)d_in[4];
    const float* lam_re = (const float*)d_in[5];
    const float* lam_im = (const float*)d_in[6];
    const float* B_re   = (const float*)d_in[7];
    const float* B_im   = (const float*)d_in[8];
    const float* C_re   = (const float*)d_in[9];
    const float* C_im   = (const float*)d_in[10];
    const float* Dskip  = (const float*)d_in[11];
    const float* lstep  = (const float*)d_in[12];
    const float* W_enc  = (const float*)d_in[13];
    const float* W_dec  = (const float*)d_in[14];
    float* out = (float*)d_out;

    char* ws = (char*)d_ws;
    size_t o = 0;
    auto carve = [&](size_t bytes) -> void* {
        o = (o + 255) & ~(size_t)255;
        void* p = ws + o;
        o += bytes;
        return p;
    };
    float*  fxh     = (float*) carve((size_t)BL * DDIM * 4);
    __bf16* fx_bf   = (__bf16*)carve((size_t)BL * DDIM * 2);
    float*  fx2_f   = (float*) carve((size_t)BL * DDIM * 4);
    __bf16* fx2_bf  = (__bf16*)carve((size_t)BL * DDIM * 2);
    float*  Bu      = (float*) carve((size_t)BL * P2 * 4);
    __bf16* xs_bf   = (__bf16*)carve((size_t)BL * P2 * 2);
    __bf16* g_bf    = (__bf16*)carve((size_t)BL * FDIM * 2);
    __bf16* Bbar_bf = (__bf16*)carve((size_t)P2 * DDIM * 2);
    __bf16* Cpack   = (__bf16*)carve((size_t)DDIM * P2 * 2);
    __bf16* Wenc_bf = (__bf16*)carve((size_t)F2 * DDIM * 2);
    __bf16* Wdec_bf = (__bf16*)carve((size_t)DDIM * FDIM * 2);
    float*  Are     = (float*) carve(PDIM * 4);
    float*  Aim     = (float*) carve(PDIM * 4);
    float*  Kre     = (float*) carve(PDIM * 4);
    float*  Kim     = (float*) carve(PDIM * 4);

    ssm_setup_kernel<<<1, 64, 0, stream>>>(lam_re, lam_im, lstep, Are, Aim, Kre, Kim);
    pack_bbar_kernel<<<(PDIM * DDIM) / 256, 256, 0, stream>>>(B_re, B_im, Kre, Kim, Bbar_bf);
    pack_c_kernel<<<(DDIM * PDIM) / 256, 256, 0, stream>>>(C_re, C_im, Cpack);
    cvt_bf16_kernel<<<((long)F2 * DDIM + 255) / 256, 256, 0, stream>>>(W_enc, Wenc_bf, (long)F2 * DDIM);
    cvt_bf16_kernel<<<((long)DDIM * FDIM + 255) / 256, 256, 0, stream>>>(W_dec, Wdec_bf, (long)DDIM * FDIM);

    ln_kernel<<<BL, 256, 0, stream>>>(x, ln1_g, ln1_b, fxh, fx_bf);
    gemm_bu_kernel<<<dim3(P2 / 128, BL / 128), 256, 0, stream>>>(fx_bf, Bbar_bf, Bu);
    scan_kernel<<<BATCH * PDIM, 256, 0, stream>>>(Bu, xs_bf, Are, Aim);
    gemm_y_kernel<<<dim3(DDIM / 128, BL / 128), 256, 0, stream>>>(xs_bf, Cpack, Dskip, fxh);
    ln_kernel<<<BL, 256, 0, stream>>>(fxh, ln2_g, ln2_b, fx2_f, fx2_bf);
    gemm_enc_kernel<<<dim3(FDIM / 128, BL / 128), 256, 0, stream>>>(fx2_bf, Wenc_bf, g_bf);
    gemm_dec_kernel<<<dim3(DDIM / 128, BL / 128), 256, 0, stream>>>(g_bf, Wdec_bf, fx2_f, out);
}
